// MTPAttention_11003706213111
// MI455X (gfx1250) — matrix operand order repaired and checked by reading, not compiled, not run on hardware
//
#include <hip/hip_runtime.h>

typedef __attribute__((ext_vector_type(16))) __bf16 bf16x16;
typedef __attribute__((ext_vector_type(8)))  float  f32x8;
typedef __attribute__((ext_vector_type(4)))  unsigned int u32x4;
typedef __attribute__((ext_vector_type(4)))  int i32x4;
typedef __attribute__((ext_vector_type(8)))  int i32x8;

namespace {
constexpr int  B_   = 2;
constexpr int  S_   = 2048;
constexpr int  HID_ = 1024;
constexpr int  NH_  = 8;
constexpr int  NKV_ = 2;
constexpr int  HD_  = 256;
constexpr int  G_   = 4;
constexpr int  ROT_ = 128;
constexpr long MR_  = (long)B_ * S_;      // 4096 token rows
constexpr float SCALE_ = 0.0625f;         // HD^-0.5
}

#if __has_builtin(__builtin_amdgcn_tensor_load_to_lds)
#define HAVE_TDM 1
#endif

// ---------------------------------------------------------------------------
// Tensor Data Mover: async 2D tile load (rows x row_elems of 2-byte data)
// from global into LDS at a static byte offset. Descriptor per CDNA5 ISA
// CDNA5 ISA. Issue once per block (TDM ignores EXEC).
// ---------------------------------------------------------------------------
__device__ __forceinline__ void tdm_load_2d(unsigned lds_byte_off, const void* gptr,
                                            unsigned tile_d0_elems, unsigned tile_d1_rows,
                                            unsigned row_stride_elems) {
#ifdef HAVE_TDM
  unsigned long ga = (unsigned long)gptr;
  u32x4 g0;
  g0[0] = 1u;                                                  // count=1, user desc
  g0[1] = lds_byte_off;                                        // lds_addr (bytes)
  g0[2] = (unsigned)(ga & 0xffffffffu);                        // global_addr[31:0]
  g0[3] = (unsigned)((ga >> 32) & 0x01ffffffu) | (2u << 30);   // addr[56:32] | type=2
  const unsigned td0 = 1u << 20, td1 = 1u << 20;               // large tensor dims (no OOB clip)
  i32x8 g1;
  g1[0] = (int)(1u << 16);                                     // data_size=1 -> 2 bytes
  g1[1] = (int)((td0 & 0xffffu) << 16);                        // tensor_dim0[15:0]
  g1[2] = (int)((td0 >> 16) | ((td1 & 0xffffu) << 16));        // td0[31:16] | td1[15:0]
  g1[3] = (int)((td1 >> 16) | (tile_d0_elems << 16));          // td1[31:16] | tile_dim0
  g1[4] = (int)tile_d1_rows;                                   // tile_dim1 (tile_dim2=0)
  g1[5] = (int)row_stride_elems;                               // tensor_dim0_stride[31:0]
  g1[6] = 0;
  g1[7] = 0;
  i32x4 g2 = {0, 0, 0, 0};
  i32x4 g3 = {0, 0, 0, 0};
#if defined(__clang_major__) && (__clang_major__ >= 23)
  i32x8 g4 = {0, 0, 0, 0, 0, 0, 0, 0};
  __builtin_amdgcn_tensor_load_to_lds(g0, g1, g2, g3, g4, 0);
#else
  __builtin_amdgcn_tensor_load_to_lds(g0, g1, g2, g3, 0);
#endif
#endif
}

// Load a 16-element bf16 fragment as two 16B LDS reads (ds_load_b128 x2).
__device__ __forceinline__ bf16x16 ld_frag(const __bf16* p0, const __bf16* p1) {
  union { bf16x16 f; uint4 u[2]; } r;
  r.u[0] = *(const uint4*)p0;
  r.u[1] = *(const uint4*)p1;
  return r.f;
}

// ---------------------------------------------------------------------------
// fp32 -> bf16 conversion (4 elems / thread)
// ---------------------------------------------------------------------------
__global__ __launch_bounds__(256)
void cvt_f32_to_bf16(const float* __restrict__ in, __bf16* __restrict__ out, long n4) {
  long i = (long)blockIdx.x * 256 + threadIdx.x;
  if (i >= n4) return;
  float4 v = ((const float4*)in)[i];
  union { __bf16 b[4]; uint2 u; } o;
  o.b[0] = (__bf16)v.x; o.b[1] = (__bf16)v.y;
  o.b[2] = (__bf16)v.z; o.b[3] = (__bf16)v.w;
  ((uint2*)out)[i] = o.u;
}

// ---------------------------------------------------------------------------
// Tiled WMMA GEMM: C(f32, MxN) = A(bf16, MxK) * B(bf16, KxN), row-major.
// Compile-time N/K -> immediate-offset stores, constant trip count.
// Block: 256 threads (8 waves), tile 128x128, K-step 32.
// ---------------------------------------------------------------------------
template <int ND, int KD>
__global__ __launch_bounds__(256)
void gemm_bf16_f32(const __bf16* __restrict__ A, const __bf16* __restrict__ Bm,
                   float* __restrict__ C) {
  __shared__ __bf16 As[128 * 32];   // row-major [m][k]
  __shared__ __bf16 Bt[128 * 32];   // transposed [n][k]
  const int tid  = threadIdx.x;
  const int lane = tid & 31, wave = tid >> 5;
  const int n0 = blockIdx.x * 128, m0 = blockIdx.y * 128;
  const int wm = (wave & 3) * 32;   // wave's row block (2 x 16)
  const int wn = (wave >> 2) * 64;  // wave's col block (4 x 16)
  const int n = lane & 15, hi = lane >> 4;

  f32x8 acc[2][4];
#pragma unroll
  for (int i = 0; i < 2; ++i)
#pragma unroll
    for (int j = 0; j < 4; ++j)
#pragma unroll
      for (int v = 0; v < 8; ++v) acc[i][j][v] = 0.0f;

  // This thread's two staging slots for A and B tiles
  const int ra0 = tid >> 2,          ca0 = (tid & 3) * 8;       // A slot 0
  const int ra1 = (tid + 256) >> 2,  ca1 = ((tid + 256) & 3) * 8;
  const int rb0 = tid >> 4,          cb0 = (tid & 15) * 8;      // B slot 0
  const int rb1 = (tid + 256) >> 4,  cb1 = ((tid + 256) & 15) * 8;

  for (int k0 = 0; k0 < KD; k0 += 32) {
    // Stage A tile 128x32 (coalesced b128 loads)
    *(uint4*)&As[ra0 * 32 + ca0] = *(const uint4*)(A + (long)(m0 + ra0) * KD + k0 + ca0);
    *(uint4*)&As[ra1 * 32 + ca1] = *(const uint4*)(A + (long)(m0 + ra1) * KD + k0 + ca1);
    // Stage B tile 32x128, transposed into Bt[n][k]
#pragma unroll
    for (int s = 0; s < 2; ++s) {
      const int rb = s ? rb1 : rb0, cb = s ? cb1 : cb0;
      uint4 u = *(const uint4*)(Bm + (long)(k0 + rb) * ND + n0 + cb);
      const __bf16* e = (const __bf16*)&u;
#pragma unroll
      for (int j = 0; j < 8; ++j) Bt[(cb + j) * 32 + rb] = e[j];
    }
    // Warm L2 for the next K-step (global_prefetch_b8)
    if (k0 + 32 < KD) {
      __builtin_prefetch(A + (long)(m0 + ra0) * KD + k0 + 32 + ca0, 0, 1);
      __builtin_prefetch(Bm + (long)(k0 + 32 + rb0) * ND + n0 + cb0, 0, 1);
    }
    __syncthreads();

    bf16x16 af[2], bfr[4];
#pragma unroll
    for (int i = 0; i < 2; ++i) {
      int row = wm + i * 16 + n;               // A-frag: lane = M row
      af[i] = ld_frag(&As[row * 32 + hi * 8], &As[row * 32 + 16 + hi * 8]);
    }
#pragma unroll
    for (int j = 0; j < 4; ++j) {
      int col = wn + j * 16 + n;               // B-frag: lane = N col
      bfr[j] = ld_frag(&Bt[col * 32 + hi * 8], &Bt[col * 32 + 16 + hi * 8]);
    }
#pragma unroll
    for (int i = 0; i < 2; ++i)
#pragma unroll
      for (int j = 0; j < 4; ++j)
        acc[i][j] = __builtin_amdgcn_wmma_f32_16x16x32_bf16(
            false, af[i], false, bfr[j], (short)0, acc[i][j], false, false);
    __syncthreads();
  }

  // C/D layout: lane = col, vgpr v = row (+8 for lanes 16-31).
  // ND is a compile-time constant -> stores use immediate offsets.
#pragma unroll
  for (int i = 0; i < 2; ++i) {
    const int row0 = m0 + wm + i * 16 + hi * 8;
#pragma unroll
    for (int j = 0; j < 4; ++j) {
      const int col = n0 + wn + j * 16 + n;
      float* cp = C + (long)row0 * ND + col;
#pragma unroll
      for (int v = 0; v < 8; ++v) cp[(long)v * ND] = acc[i][j][v];
    }
  }
}

// ---------------------------------------------------------------------------
// Per-row RMSNorm(256) * (1+w) + RoPE(first 128 dims), fp32 in -> bf16 out.
// One wave per (b,s,head) row; 8 elems/lane; shfl reductions only.
// ---------------------------------------------------------------------------
__global__ __launch_bounds__(256)
void norm_rope_kernel(const float* __restrict__ in, int row_stride, int heads,
                      const float* __restrict__ w,
                      const float* __restrict__ cosb, const float* __restrict__ sinb,
                      __bf16* __restrict__ out) {
  const int lane = threadIdx.x & 31, wave = threadIdx.x >> 5;
  const long row = (long)blockIdx.x * 8 + wave;
  const long tok = row / heads;                 // (b*S + s)
  const float* x = in + row * row_stride;

  float v[8]; float ss = 0.0f;
#pragma unroll
  for (int j = 0; j < 8; ++j) { v[j] = x[lane * 8 + j]; ss += v[j] * v[j]; }
#pragma unroll
  for (int o = 16; o > 0; o >>= 1) ss += __shfl_xor(ss, o);
  const float rs = rsqrtf(ss * (1.0f / 256.0f) + 1e-6f);

  float nv[8];
#pragma unroll
  for (int j = 0; j < 8; ++j) nv[j] = v[j] * rs * (1.0f + w[lane * 8 + j]);

  union { __bf16 b[8]; uint4 u; } o8;
#pragma unroll
  for (int j = 0; j < 8; ++j) {
    const int i = lane * 8 + j;
    const float rot = __shfl_xor(nv[j], 8);     // partner at i +/- 64
    float r = nv[j];
    if (i < ROT_) {
      const float c = cosb[tok * ROT_ + i];
      const float s = sinb[tok * ROT_ + i];
      r = nv[j] * c + ((i < ROT_ / 2) ? -rot : rot) * s;
    }
    o8.b[j] = (__bf16)r;
  }
  *(uint4*)(out + row * HD_ + lane * 8) = o8.u;
}

// ---------------------------------------------------------------------------
// Flash attention (causal, GQA). Block = (128 q rows, head h, batch b),
// 8 waves x 16 q rows. Computes S^T = K*Q^T so each lane owns one query
// column: online softmax stats need only shfl_xor(16). O^T accumulated via
// WMMA(A = V^T tile from LDS, B = P^T built in-register from S^T tiles).
// K tile staged by the Tensor Data Mover (async tensor load + TENSORcnt).
// ---------------------------------------------------------------------------
__global__ __launch_bounds__(256)
void attn_kernel(const __bf16* __restrict__ Qp, const __bf16* __restrict__ Kp,
                 const __bf16* __restrict__ Vp, float* __restrict__ Op) {
  // One merged allocation so LDS byte offsets are known for the TDM descriptor.
  __shared__ __bf16 smem[128 * HD_ + 32 * HD_ + HD_ * 32];  // 96 KB
  __bf16* QL = smem;                      // [128][256]   byte off 0
  __bf16* KL = smem + 128 * HD_;          // [32][256]    byte off 65536
  __bf16* VT = smem + 128 * HD_ + 32 * HD_;  // [256][32] byte off 81920
  constexpr unsigned KL_OFF = 128u * HD_ * 2u;

  const int tid  = threadIdx.x;
  const int lane = tid & 31, wave = tid >> 5;
  const int qb = blockIdx.x * 128;
  const int h  = blockIdx.y;
  const int b  = blockIdx.z;
  const int kv = h / G_;
  const int n = lane & 15, hi = lane >> 4;

  { // stage Q block (128 rows x 256 d)
    const __bf16* qsrc = Qp + (((long)b * S_ + qb) * NH_ + h) * HD_;
#pragma unroll
    for (int i = 0; i < 16; ++i) {
      int idx = i * 256 + tid;
      int r = idx >> 5, c = (idx & 31) * 8;
      *(uint4*)&QL[r * HD_ + c] = *(const uint4*)(qsrc + (long)r * NH_ * HD_ + c);
    }
  }

  const int qrow  = wave * 16 + n;   // lane's query row within block
  const int qglob = qb + qrow;

  f32x8 oacc[16];                    // O^T: 16 d-tiles x (16d x 16q)
#pragma unroll
  for (int t = 0; t < 16; ++t)
#pragma unroll
    for (int v = 0; v < 8; ++v) oacc[t][v] = 0.0f;
  float m_run = -1e30f, l_run = 0.0f;

  const __bf16* kbase0 = Kp + ((long)b * S_ * NKV_ + kv) * HD_;
  const __bf16* vbase0 = Vp + ((long)b * S_ * NKV_ + kv) * HD_;

  const int nkt = (qb + 128) / 32;
  for (int kt = 0; kt < nkt; ++kt) {
    const int kb = kt * 32;
    const __bf16* ksrc = kbase0 + (long)kb * NKV_ * HD_;
    const __bf16* vsrc = vbase0 + (long)kb * NKV_ * HD_;

#ifdef HAVE_TDM
    if (wave == 0) {   // one TDM issue per block; EXEC is ignored by TDM
      tdm_load_2d(KL_OFF, ksrc, HD_, 32, NKV_ * HD_);
      __builtin_amdgcn_s_wait_tensorcnt(0);
    }
#else
#pragma unroll
    for (int i = 0; i < 4; ++i) {
      int idx = i * 256 + tid;
      int r = idx >> 5, c = (idx & 31) * 8;
      *(uint4*)&KL[r * HD_ + c] = *(const uint4*)(ksrc + (long)r * NKV_ * HD_ + c);
    }
#endif
    { // stage V tile transposed: VT[d][k]  (the scatter *is* the transpose)
#pragma unroll
      for (int i = 0; i < 4; ++i) {
        int idx = i * 256 + tid;
        int r = idx >> 5, c = (idx & 31) * 8;   // r = key row, c = d base
        uint4 u = *(const uint4*)(vsrc + (long)r * NKV_ * HD_ + c);
        const __bf16* e = (const __bf16*)&u;
#pragma unroll
        for (int j = 0; j < 8; ++j) VT[(c + j) * 32 + r] = e[j];
      }
    }
    if (kt + 1 < nkt) {  // warm L2 for next tiles
      __builtin_prefetch(ksrc + (long)32 * NKV_ * HD_ + lane * 64, 0, 1);
      __builtin_prefetch(vsrc + (long)32 * NKV_ * HD_ + lane * 64, 0, 1);
    }
    __syncthreads();

    // S^T (32k x 16q) as two 16x16 tiles: A = K rows, B = Q^T
    f32x8 st0, st1;
#pragma unroll
    for (int v = 0; v < 8; ++v) { st0[v] = 0.0f; st1[v] = 0.0f; }
#pragma unroll
    for (int dc = 0; dc < 8; ++dc) {
      const int d0 = dc * 32;
      bf16x16 bq = ld_frag(&QL[qrow * HD_ + d0 + hi * 8],
                           &QL[qrow * HD_ + d0 + 16 + hi * 8]);
      bf16x16 a0 = ld_frag(&KL[n * HD_ + d0 + hi * 8],
                           &KL[n * HD_ + d0 + 16 + hi * 8]);
      bf16x16 a1 = ld_frag(&KL[(n + 16) * HD_ + d0 + hi * 8],
                           &KL[(n + 16) * HD_ + d0 + 16 + hi * 8]);
      st0 = __builtin_amdgcn_wmma_f32_16x16x32_bf16(false, a0, false, bq, (short)0, st0, false, false);
      st1 = __builtin_amdgcn_wmma_f32_16x16x32_bf16(false, a1, false, bq, (short)0, st1, false, false);
    }

    // Online softmax: lane owns query column qglob; keys split across
    // lane and lane^16 -> one shfl_xor(16) for stats.
    float sc[2][8];
    float mloc = -1e30f;
#pragma unroll
    for (int t = 0; t < 2; ++t)
#pragma unroll
      for (int v = 0; v < 8; ++v) {
        float s = (t ? st1[v] : st0[v]) * SCALE_;
        const int kg = kb + t * 16 + hi * 8 + v;
        if (kg > qglob) s = -1e30f;             // causal mask
        sc[t][v] = s;
        mloc = fmaxf(mloc, s);
      }
    mloc = fmaxf(mloc, __shfl_xor(mloc, 16));
    const float mnew  = fmaxf(m_run, mloc);
    const float alpha = __expf(m_run - mnew);
    m_run = mnew;

    float p[2][8]; float psum = 0.0f;
#pragma unroll
    for (int t = 0; t < 2; ++t)
#pragma unroll
      for (int v = 0; v < 8; ++v) { p[t][v] = __expf(sc[t][v] - mnew); psum += p[t][v]; }
    l_run = l_run * alpha + psum;

#pragma unroll
    for (int t = 0; t < 16; ++t)
#pragma unroll
      for (int v = 0; v < 8; ++v) oacc[t][v] *= alpha;

    // Build P^T B-fragment (32k x 16q) from the two S^T C-layout tiles.
    float pp[2][8];
#pragma unroll
    for (int t = 0; t < 2; ++t)
#pragma unroll
      for (int v = 0; v < 8; ++v) pp[t][v] = __shfl_xor(p[t][v], 16);
    union { bf16x16 f; __bf16 e[16]; } pu;
#pragma unroll
    for (int v = 0; v < 8; ++v) {
      pu.e[v]     = (__bf16)(hi ? pp[1][v] : p[0][v]);   // k = hi*16 + v
      pu.e[v + 8] = (__bf16)(hi ? p[1][v]  : pp[0][v]);  // k = hi*16 + 8 + v
    }

    // O^T += V^T * P^T : 16 d-tiles, each A = V^T (16d x 32k)
#pragma unroll
    for (int t = 0; t < 16; ++t) {
      bf16x16 av = ld_frag(&VT[(t * 16 + n) * 32 + hi * 16],
                           &VT[(t * 16 + n) * 32 + hi * 16 + 8]);
      oacc[t] = __builtin_amdgcn_wmma_f32_16x16x32_bf16(
          false, av, false, pu.f, (short)0, oacc[t], false, false);
    }
    __syncthreads();
  }

  const float ltot = l_run + __shfl_xor(l_run, 16);
  const float inv  = 1.0f / ltot;
  float* ob = Op + (((long)b * S_ + qglob) * NH_ + h) * HD_;
#pragma unroll
  for (int t = 0; t < 16; ++t) {
    const int d0 = t * 16 + hi * 8;            // lane holds 8 consecutive d
    float4 lo  = make_float4(oacc[t][0] * inv, oacc[t][1] * inv, oacc[t][2] * inv, oacc[t][3] * inv);
    float4 hi4 = make_float4(oacc[t][4] * inv, oacc[t][5] * inv, oacc[t][6] * inv, oacc[t][7] * inv);
    *(float4*)(ob + d0)     = lo;
    *(float4*)(ob + d0 + 4) = hi4;
  }
}

// ---------------------------------------------------------------------------
// Gating: out_bf16 = attn_f32 * sigmoid(gate) ; gate lives in qg[..., 256:512]
// ---------------------------------------------------------------------------
__global__ __launch_bounds__(256)
void gate_kernel(const float* __restrict__ attn, const float* __restrict__ qg,
                 __bf16* __restrict__ out) {
  long i4 = (long)blockIdx.x * 256 + threadIdx.x;
  long i  = i4 * 4;
  long rh = i >> 8;                  // (b,s,head) row
  int  d  = (int)(i & 255);
  float4 a = *(const float4*)(attn + i);
  const float* g = qg + rh * 512 + 256 + d;
  float av[4] = {a.x, a.y, a.z, a.w};
  union { __bf16 b[4]; uint2 u; } o;
#pragma unroll
  for (int j = 0; j < 4; ++j) {
    float sg = 1.0f / (1.0f + __expf(-g[j]));
    o.b[j] = (__bf16)(av[j] * sg);
  }
  *(uint2*)(out + i) = o.u;
}

// ---------------------------------------------------------------------------
extern "C" void kernel_launch(void* const* d_in, const int* /*in_sizes*/, int /*n_in*/,
                              void* d_out, int /*out_size*/, void* d_ws, size_t /*ws_size*/,
                              hipStream_t stream) {
  const float* hs   = (const float*)d_in[0];
  const float* cosb = (const float*)d_in[1];
  const float* sinb = (const float*)d_in[2];
  const float* Wq   = (const float*)d_in[3];
  const float* Wk   = (const float*)d_in[4];
  const float* Wv   = (const float*)d_in[5];
  const float* Wo   = (const float*)d_in[6];
  const float* qnw  = (const float*)d_in[7];
  const float* knw  = (const float*)d_in[8];
  float* out = (float*)d_out;

  char* ws = (char*)d_ws;
  size_t off = 0;
  auto alloc = [&](size_t bytes) -> void* {
    void* p = ws + off;
    off += (bytes + 255) & ~(size_t)255;
    return p;
  };
  __bf16* hsb = (__bf16*)alloc(MR_ * HID_ * 2);           // hidden, bf16
  __bf16* wqb = (__bf16*)alloc((long)HID_ * 4096 * 2);
  __bf16* wkb = (__bf16*)alloc((long)HID_ * 512 * 2);
  __bf16* wvb = (__bf16*)alloc((long)HID_ * 512 * 2);
  __bf16* wob = (__bf16*)alloc((long)2048 * HID_ * 2);
  float*  qg  = (float*) alloc(MR_ * 4096 * 4);           // q+gate, f32
  float*  kf  = (float*) alloc(MR_ * 512 * 4);
  float*  vf  = (float*) alloc(MR_ * 512 * 4);
  __bf16* Qb  = (__bf16*)alloc(MR_ * 2048 * 2);           // normed+roped q
  __bf16* Kb  = (__bf16*)alloc(MR_ * 512 * 2);
  __bf16* Vb  = (__bf16*)alloc(MR_ * 512 * 2);
  float*  att = (float*) alloc(MR_ * 2048 * 4);           // attention out
  __bf16* gtd = (__bf16*)alloc(MR_ * 2048 * 2);           // gated, bf16

  auto cvt = [&](const float* s, __bf16* d, long nels) {
    long n4 = nels / 4;
    cvt_f32_to_bf16<<<dim3((unsigned)((n4 + 255) / 256)), dim3(256), 0, stream>>>(s, d, n4);
  };
  cvt(hs, hsb, MR_ * HID_);
  cvt(Wq, wqb, (long)HID_ * 4096);
  cvt(Wk, wkb, (long)HID_ * 512);
  cvt(Wv, wvb, (long)HID_ * 512);
  cvt(Wo, wob, (long)2048 * HID_);

  // Projections (compile-time N/K)
  gemm_bf16_f32<4096, 1024><<<dim3(4096 / 128, 32), dim3(256), 0, stream>>>(hsb, wqb, qg);
  gemm_bf16_f32<512, 1024><<<dim3(512 / 128, 32), dim3(256), 0, stream>>>(hsb, wkb, kf);
  gemm_bf16_f32<512, 1024><<<dim3(512 / 128, 32), dim3(256), 0, stream>>>(hsb, wvb, vf);

  // RMSNorm + RoPE (q: 32768 rows, stride 512 ; k: 8192 rows, stride 256)
  norm_rope_kernel<<<dim3(4096), dim3(256), 0, stream>>>(qg, 512, NH_, qnw, cosb, sinb, Qb);
  norm_rope_kernel<<<dim3(1024), dim3(256), 0, stream>>>(kf, 256, NKV_, knw, cosb, sinb, Kb);
  cvt(vf, Vb, MR_ * 512);

  attn_kernel<<<dim3(S_ / 128, NH_, B_), dim3(256), 0, stream>>>(Qb, Kb, Vb, att);

  gate_kernel<<<dim3((unsigned)(MR_ * 2048 / 4 / 256)), dim3(256), 0, stream>>>(att, qg, gtd);

  gemm_bf16_f32<1024, 2048><<<dim3(1024 / 128, 32), dim3(256), 0, stream>>>(gtd, wob, out);
}
